// TinySelfAttention_68135361184085
// MI455X (gfx1250) — matrix operand order repaired and checked by reading, not compiled, not run on hardware
//
#include <hip/hip_runtime.h>

// ---------- CDNA5 WMMA / TDM types ----------
typedef __attribute__((ext_vector_type(16))) __bf16 v16bf;
typedef __attribute__((ext_vector_type(8)))  __bf16 v8bf;
typedef __attribute__((ext_vector_type(8)))  float  v8f;
typedef __attribute__((ext_vector_type(4)))  unsigned int v4u;
typedef __attribute__((ext_vector_type(8)))  int v8i;
typedef __attribute__((ext_vector_type(4)))  int v4i;

#define WMMA_BF16(A, B, C) \
  __builtin_amdgcn_wmma_f32_16x16x32_bf16(false, (A), false, (B), (short)0, (C), false, false)

#if defined(__has_builtin)
#if __has_builtin(__builtin_amdgcn_tensor_load_to_lds) && \
    __has_builtin(__builtin_amdgcn_s_wait_tensorcnt)
#define USE_TDM 1
#else
#define USE_TDM 0
#endif
#else
#define USE_TDM 0
#endif

// Load a 16-element bf16 fragment from two 16-byte chunks (ds_load_b128 pairs).
__device__ __forceinline__ v16bf load_frag(const __bf16* p0, const __bf16* p1) {
  v8bf lo = *(const v8bf*)p0;
  v8bf hi = *(const v8bf*)p1;
  return __builtin_shufflevector(lo, hi, 0, 1, 2, 3, 4, 5, 6, 7,
                                 8, 9, 10, 11, 12, 13, 14, 15);
}

#if USE_TDM
// Issue a 2D bf16 TDM load (tile_d1 rows x tile_d0 elems) into LDS with row padding.
// pad codes per ISA D# group1: interval 0=2DW..7=256DW ; amount 0=1DW..127=128DW
__device__ __forceinline__ void tdm_load_2d(unsigned lds_off, const __bf16* gptr,
                                            unsigned tile_d0, unsigned tile_d1,
                                            unsigned tensor_d0, unsigned tensor_d1,
                                            unsigned d0_stride, unsigned pad_interval,
                                            unsigned pad_amount) {
  unsigned long long ga = (unsigned long long)(uintptr_t)gptr;
  v4u g0;
  g0.x = 1u;                                            // count=1, user descriptor
  g0.y = lds_off;                                       // lds_addr [63:32]
  g0.z = (unsigned)(ga & 0xFFFFFFFFu);                  // global_addr [95:64]
  g0.w = (unsigned)((ga >> 32) & 0x1FFFFFFu) | (2u << 30);  // addr[56:32] | type=2
  v8i g1;
  g1[0] = (int)((1u << 16) |                            // data_size = 2 bytes
                (1u << 20) |                            // pad_enable
                (pad_interval << 22) | (pad_amount << 25));
  g1[1] = (int)((tensor_d0 & 0xFFFFu) << 16);           // atomic_bar=0, dim0[15:0]
  g1[2] = (int)(((tensor_d0 >> 16) & 0xFFFFu) | ((tensor_d1 & 0xFFFFu) << 16));
  g1[3] = (int)(((tensor_d1 >> 16) & 0xFFFFu) | ((tile_d0 & 0xFFFFu) << 16));
  g1[4] = (int)(tile_d1 & 0xFFFFu);                     // tile_dim1 ; tile_dim2=0
  g1[5] = (int)d0_stride;                               // dim0 stride (low 32)
  g1[6] = 0;                                            // stride hi, dim1_stride lo
  g1[7] = 0;
  v4i gz = {0, 0, 0, 0};
  v8i gz8 = {0, 0, 0, 0, 0, 0, 0, 0};
  __builtin_amdgcn_tensor_load_to_lds(g0, g1, gz, gz, gz8, 0);
}
#endif

// ---------- conversion kernels (one-shot) ----------
__global__ void cvt_f32_to_bf16(const float* __restrict__ in, __bf16* __restrict__ out, int n) {
  int i = blockIdx.x * 256 + threadIdx.x;
  if (i < n) out[i] = (__bf16)in[i];
}

// in: [K][N] row-major fp32  ->  out: [N][K] row-major bf16
__global__ void cvt_transpose_bf16(const float* __restrict__ in, __bf16* __restrict__ out,
                                   int N, int K) {
  int i = blockIdx.x * 256 + threadIdx.x;
  if (i < N * K) {
    int n = i / K;
    int k = i - n * K;
    out[i] = (__bf16)in[(size_t)k * N + n];
  }
}

// ---------- tiled bf16 GEMM with WMMA, double-buffered LDS ----------
// C[M,N] = A[M,K] * Bt[N,K]^T + bias ; EPI 0: scatter to Q/K/Vt bf16, EPI 1: fp32 out
template <int EPI>
__global__ __launch_bounds__(256) void gemm_bf16_wmma(
    const __bf16* __restrict__ A, const __bf16* __restrict__ Bt,
    const float* __restrict__ bias, int M, int N, int K,
    __bf16* __restrict__ q_out, __bf16* __restrict__ k_out,
    __bf16* __restrict__ vt_out, float* __restrict__ f_out) {
  constexpr int BM = 128, BN = 128, BK = 32, LS = BK + 16;  // 16B-aligned row stride
  __shared__ __align__(16) __bf16 As[2][BM * LS];
  __shared__ __align__(16) __bf16 Bs[2][BN * LS];

  const int tid  = threadIdx.x;
  const int wave = tid >> 5;
  const int lane = tid & 31;
  const int half = lane >> 4;
  const int l16  = lane & 15;
  const int wm   = wave & 3;   // 4 wave-rows * 32 rows
  const int wn   = wave >> 2;  // 2 wave-cols * 64 cols
  const int bm0  = blockIdx.x * BM;
  const int bn0  = blockIdx.y * BN;

  // each thread owns two 16B chunks per operand tile
  const int r0 = tid >> 2, k0 = (tid & 3) * 8;
  const int r1 = (tid + 256) >> 2, k1 = ((tid + 256) & 3) * 8;

  auto ldA = [&](int kb, int r, int k) { return *(const v8bf*)&A[(size_t)(bm0 + r) * K + kb + k]; };
  auto ldB = [&](int kb, int r, int k) { return *(const v8bf*)&Bt[(size_t)(bn0 + r) * K + kb + k]; };

  v8f acc[2][4] = {};

  v8bf ra0 = ldA(0, r0, k0), ra1 = ldA(0, r1, k1);
  v8bf rb0 = ldB(0, r0, k0), rb1 = ldB(0, r1, k1);
  *(v8bf*)&As[0][r0 * LS + k0] = ra0;
  *(v8bf*)&As[0][r1 * LS + k1] = ra1;
  *(v8bf*)&Bs[0][r0 * LS + k0] = rb0;
  *(v8bf*)&Bs[0][r1 * LS + k1] = rb1;
  __syncthreads();

  int cur = 0;
  for (int kb = 0; kb < K; kb += BK) {
    const bool nxt = (kb + BK) < K;
    if (nxt) {  // prefetch next tile while current tile computes
      ra0 = ldA(kb + BK, r0, k0); ra1 = ldA(kb + BK, r1, k1);
      rb0 = ldB(kb + BK, r0, k0); rb1 = ldB(kb + BK, r1, k1);
    }

    v16bf af[2], bfr[4];
#pragma unroll
    for (int mt = 0; mt < 2; ++mt) {
      int row = (wm * 32 + mt * 16 + l16) * LS;
      af[mt] = load_frag(&As[cur][row + half * 8], &As[cur][row + 16 + half * 8]);
    }
#pragma unroll
    for (int nt = 0; nt < 4; ++nt) {
      int row = (wn * 64 + nt * 16 + l16) * LS;
      bfr[nt] = load_frag(&Bs[cur][row + half * 8], &Bs[cur][row + 16 + half * 8]);
    }
#pragma unroll
    for (int mt = 0; mt < 2; ++mt)
#pragma unroll
      for (int nt = 0; nt < 4; ++nt) acc[mt][nt] = WMMA_BF16(af[mt], bfr[nt], acc[mt][nt]);

    if (nxt) {
      *(v8bf*)&As[cur ^ 1][r0 * LS + k0] = ra0;
      *(v8bf*)&As[cur ^ 1][r1 * LS + k1] = ra1;
      *(v8bf*)&Bs[cur ^ 1][r0 * LS + k0] = rb0;
      *(v8bf*)&Bs[cur ^ 1][r1 * LS + k1] = rb1;
    }
    __syncthreads();
    cur ^= 1;
  }

  // epilogue: C layout -> VGPR r holds (M = half*8 + r, N = lane&15)
#pragma unroll
  for (int mt = 0; mt < 2; ++mt)
#pragma unroll
    for (int nt = 0; nt < 4; ++nt) {
      int gn = bn0 + wn * 64 + nt * 16 + l16;
      float bv = bias[gn];
#pragma unroll
      for (int r = 0; r < 8; ++r) {
        int gm = bm0 + wm * 32 + mt * 16 + half * 8 + r;
        float c = acc[mt][nt][r] + bv;
        if (EPI == 0) {
          int b = gm >> 11, s = gm & 2047;
          int sel = gn >> 11, rem = gn & 2047;
          int h = rem >> 7, hd = rem & 127;
          size_t qk = ((size_t)(b * 16 + h) * 2048 + s) * 128 + hd;
          if (sel == 0)      q_out[qk] = (__bf16)c;
          else if (sel == 1) k_out[qk] = (__bf16)c;
          else vt_out[((size_t)(b * 16 + h) * 128 + hd) * 2048 + s] = (__bf16)c;
        } else {
          f_out[(size_t)gm * N + gn] = c;
        }
      }
    }
}

// ---------- flash attention (causal) ----------
// Q/K: [bh][s=2048][hd=128] bf16 ; Vt: [bh][hd=128][s=2048] bf16 ; O: [b*s][d=2048] bf16
__global__ __launch_bounds__(128) void attn_wmma(
    const __bf16* __restrict__ Q, const __bf16* __restrict__ Kmat,
    const __bf16* __restrict__ Vt, __bf16* __restrict__ O) {
  constexpr int KS = 136, VS = 40, PS = 40;  // padded, 16B-aligned rows
  __shared__ __align__(16) __bf16 Ks[32 * KS];
  __shared__ __align__(16) __bf16 Vs[128 * VS];
  __shared__ __align__(16) __bf16 Ps[4 * 16 * PS];

  const int tid = threadIdx.x, wave = tid >> 5, lane = tid & 31;
  const int half = lane >> 4, l16 = lane & 15;
  const int qb = blockIdx.x;  // 32 q-blocks of 64 rows
  const int bh = blockIdx.y;  // 32 (b,h) pairs
  const size_t qkb = (size_t)bh * 2048 * 128;
  const size_t vtb = (size_t)bh * 128 * 2048;
  const int q0 = qb * 64 + wave * 16;
  const float scale = 0.08838834764831845f;  // 1/sqrt(128)

  // Q fragments for the whole pass (A-matrix layout, hd = K dim, 4 x K32)
  v16bf qf[4];
  {
    const __bf16* qrow = Q + qkb + (size_t)(q0 + l16) * 128;
#pragma unroll
    for (int f = 0; f < 4; ++f)
      qf[f] = load_frag(qrow + f * 32 + half * 8, qrow + f * 32 + 16 + half * 8);
  }

  v8f oacc[8] = {};
  float rmax[8], rsum[8];
#pragma unroll
  for (int r = 0; r < 8; ++r) { rmax[r] = -1e30f; rsum[r] = 0.0f; }

  const int kend = qb * 64 + 64;  // causal bound
  for (int kc = 0; kc < kend; kc += 32) {
#if USE_TDM
    // Tensor Data Mover: DMA K chunk (32x128) and V chunk (128x32) into padded LDS
    if (tid < 32) {
      tdm_load_2d((unsigned)(uintptr_t)&Ks[0], Kmat + qkb + (size_t)kc * 128,
                  /*tile_d0=*/128, /*tile_d1=*/32,
                  /*tensor_d0=*/128, /*tensor_d1=*/(unsigned)(2048 - kc),
                  /*d0_stride=*/128,
                  /*pad_interval=*/5 /*64 DW = 1 row*/, /*pad_amount=*/3 /*4 DW*/);
      tdm_load_2d((unsigned)(uintptr_t)&Vs[0], Vt + vtb + kc,
                  /*tile_d0=*/32, /*tile_d1=*/128,
                  /*tensor_d0=*/(unsigned)(2048 - kc), /*tensor_d1=*/128,
                  /*d0_stride=*/2048,
                  /*pad_interval=*/3 /*16 DW = 1 row*/, /*pad_amount=*/3 /*4 DW*/);
      __builtin_amdgcn_s_wait_tensorcnt(0);
    }
#else
#pragma unroll
    for (int i = 0; i < 4; ++i) {
      int c = tid + i * 128;
      int kr = c >> 4, ko = (c & 15) * 8;
      *(v8bf*)&Ks[kr * KS + ko] = *(const v8bf*)&Kmat[qkb + (size_t)(kc + kr) * 128 + ko];
      int vr = c >> 2, vo = (c & 3) * 8;
      *(v8bf*)&Vs[vr * VS + vo] = *(const v8bf*)&Vt[vtb + (size_t)vr * 2048 + kc + vo];
    }
#endif
    __syncthreads();

    // S = Q K^T (scaled later): two 16-key tiles, K-dim chained over hd
    v8f s[2] = {};
#pragma unroll
    for (int nt = 0; nt < 2; ++nt)
#pragma unroll
      for (int f = 0; f < 4; ++f) {
        int row = (nt * 16 + l16) * KS + f * 32 + half * 8;
        v16bf kf = load_frag(&Ks[row], &Ks[row + 16]);
        s[nt] = WMMA_BF16(qf[f], kf, s[nt]);
      }

    // online softmax: rows live in 16-lane halves (M = half*8 + r)
    float pr[2][8], alpha[8];
#pragma unroll
    for (int r = 0; r < 8; ++r) {
      int qg = q0 + half * 8 + r;
      float mloc = -1e30f;
#pragma unroll
      for (int nt = 0; nt < 2; ++nt) {
        int kg = kc + nt * 16 + l16;
        float v = s[nt][r] * scale;
        if (kg > qg) v = -1e30f;
        s[nt][r] = v;
        mloc = fmaxf(mloc, v);
      }
#pragma unroll
      for (int m = 1; m < 16; m <<= 1) mloc = fmaxf(mloc, __shfl_xor(mloc, m, 32));
      float mnew = fmaxf(rmax[r], mloc);
      alpha[r] = __expf(rmax[r] - mnew);
      rmax[r] = mnew;
      float lsum = 0.0f;
#pragma unroll
      for (int nt = 0; nt < 2; ++nt) {
        float p = __expf(s[nt][r] - mnew);
        pr[nt][r] = p;
        lsum += p;
      }
#pragma unroll
      for (int m = 1; m < 16; m <<= 1) lsum += __shfl_xor(lsum, m, 32);
      rsum[r] = rsum[r] * alpha[r] + lsum;
    }
#pragma unroll
    for (int nt = 0; nt < 8; ++nt)
#pragma unroll
      for (int r = 0; r < 8; ++r) oacc[nt][r] *= alpha[r];

    // P: C-layout -> LDS -> A-layout fragment (per-wave private region)
#pragma unroll
    for (int nt = 0; nt < 2; ++nt)
#pragma unroll
      for (int r = 0; r < 8; ++r)
        Ps[(wave * 16 + half * 8 + r) * PS + nt * 16 + l16] = (__bf16)pr[nt][r];
    __syncthreads();

    int prow = (wave * 16 + l16) * PS;
    v16bf pf = load_frag(&Ps[prow + half * 8], &Ps[prow + 16 + half * 8]);
#pragma unroll
    for (int nt = 0; nt < 8; ++nt) {
      int vrow = (nt * 16 + l16) * VS + half * 8;
      v16bf vf = load_frag(&Vs[vrow], &Vs[vrow + 16]);
      oacc[nt] = WMMA_BF16(pf, vf, oacc[nt]);
    }
    __syncthreads();
  }

  // normalize and store O as [b*s][h*128+hd] bf16
  const int b = bh >> 4, h = bh & 15;
#pragma unroll
  for (int r = 0; r < 8; ++r) {
    float inv = 1.0f / rsum[r];
    int qg = q0 + half * 8 + r;
    size_t row = (size_t)(b * 2048 + qg) * 2048 + h * 128;
#pragma unroll
    for (int nt = 0; nt < 8; ++nt) O[row + nt * 16 + l16] = (__bf16)(oacc[nt][r] * inv);
  }
}

// ---------- host launch ----------
extern "C" void kernel_launch(void* const* d_in, const int* in_sizes, int n_in,
                              void* d_out, int out_size, void* d_ws, size_t ws_size,
                              hipStream_t stream) {
  const float* x     = (const float*)d_in[0];
  const float* W_qkv = (const float*)d_in[1];
  const float* b_qkv = (const float*)d_in[2];
  const float* W_out = (const float*)d_in[3];
  const float* b_out = (const float*)d_in[4];

  char* ws = (char*)d_ws;
  __bf16* xb  = (__bf16*)(ws + 0);                    // 4096x2048
  __bf16* wqt = (__bf16*)(ws + 16777216ull);          // 6144x2048 (W_qkv^T)
  __bf16* wot = (__bf16*)(ws + 41943040ull);          // 2048x2048 (W_out^T)
  __bf16* Qb  = (__bf16*)(ws + 50331648ull);          // [32][2048][128]
  __bf16* Kb  = (__bf16*)(ws + 67108864ull);          // [32][2048][128]
  __bf16* Vtb = (__bf16*)(ws + 83886080ull);          // [32][128][2048]
  __bf16* Ob  = (__bf16*)(ws + 100663296ull);         // 4096x2048

  int nx = 4096 * 2048;
  cvt_f32_to_bf16<<<(nx + 255) / 256, 256, 0, stream>>>(x, xb, nx);
  int nq = 6144 * 2048;
  cvt_transpose_bf16<<<(nq + 255) / 256, 256, 0, stream>>>(W_qkv, wqt, 6144, 2048);
  int no = 2048 * 2048;
  cvt_transpose_bf16<<<(no + 255) / 256, 256, 0, stream>>>(W_out, wot, 2048, 2048);

  // QKV projection: [4096,2048] x [2048,6144] -> scatter Q/K/Vt
  gemm_bf16_wmma<0><<<dim3(32, 48), 256, 0, stream>>>(
      xb, wqt, b_qkv, 4096, 6144, 2048, Qb, Kb, Vtb, nullptr);

  // causal flash attention
  attn_wmma<<<dim3(32, 32), 128, 0, stream>>>(Qb, Kb, Vtb, Ob);

  // output projection: [4096,2048] x [2048,2048] -> fp32 d_out
  gemm_bf16_wmma<1><<<dim3(32, 16), 256, 0, stream>>>(
      Ob, wot, b_out, 4096, 2048, 2048, nullptr, nullptr, nullptr, (float*)d_out);
}
